// MulitLatentAttention_26096221290512
// MI455X (gfx1250) — matrix operand order repaired and checked by reading, not compiled, not run on hardware
//
#include <hip/hip_runtime.h>
#include <stdint.h>

typedef unsigned short bf16_t;                                   // raw bf16 bits
typedef __attribute__((ext_vector_type(16))) __bf16 v16bf;
typedef __attribute__((ext_vector_type(8)))  float  v8f;
typedef int v4i __attribute__((vector_size(16)));                // int4 vector

static constexpr int Bc = 2, Sc = 2048, Dc = 2048, Hc = 16;
static constexpr int QLORA = 768, KVLORA = 512;
static constexpr int NOPE = 128, ROPE = 64, VD = 128;
static constexpr int QHD  = NOPE + ROPE;   // 192
static constexpr int KVUP = NOPE + VD;     // 256 per head
static constexpr int MROWS = Bc * Sc;      // 4096

// ---- CDNA5 async global->LDS copy (ASYNCcnt path), with sync fallback ------
#if defined(__has_builtin)
#if __has_builtin(__builtin_amdgcn_global_load_async_to_lds_b128)
#define HAVE_ASYNC_LDS 1
#endif
#endif
#ifndef HAVE_ASYNC_LDS
#define HAVE_ASYNC_LDS 0
#endif

#if HAVE_ASYNC_LDS
typedef __attribute__((address_space(1))) v4i* g_v4i_ptr;   // global int4*
typedef __attribute__((address_space(3))) v4i* l_v4i_ptr;   // LDS int4*
__device__ __forceinline__ void async_g2l_b128(const void* g, void* l) {
  __builtin_amdgcn_global_load_async_to_lds_b128((g_v4i_ptr)g, (l_v4i_ptr)l, 0,
                                                 0);
}
__device__ __forceinline__ void wait_async0() {
#if __has_builtin(__builtin_amdgcn_s_wait_asynccnt)
  __builtin_amdgcn_s_wait_asynccnt(0);
#else
  asm volatile("s_wait_asynccnt 0x0" ::: "memory");
#endif
}
#else
__device__ __forceinline__ void wait_async0() {}
#endif

__device__ __forceinline__ bf16_t f32_to_bf16(float f) {
  unsigned u = __builtin_bit_cast(unsigned, f);
  unsigned r = u + 0x7FFFu + ((u >> 16) & 1u);   // round-to-nearest-even
  return (bf16_t)(r >> 16);
}

__device__ __forceinline__ v8f wmma_bf16(v16bf a, v16bf b, v8f c) {
  return __builtin_amdgcn_wmma_f32_16x16x32_bf16(false, a, false, b, (short)0, c,
                                                 false, false);
}

// A-fragment (16x32 bf16): lane(lr,lh) holds row=lr, K in [8*lh,8*lh+8) U [16+8*lh,+8)
__device__ __forceinline__ v16bf load_afrag(const bf16_t* rowp, int lh) {
  union { uint4 q[2]; v16bf v; } u;
  u.q[0] = *(const uint4*)(rowp + 8 * lh);
  u.q[1] = *(const uint4*)(rowp + 16 + 8 * lh);
  return u.v;
}

// B-fragment (32x16 bf16): lane(lr,lh) holds col=lr, K in [8*lh,8*lh+8) U [16+8*lh,+8)
// colp points at the K-contiguous storage of one column (transposed tile).
__device__ __forceinline__ v16bf load_bfrag(const bf16_t* colp, int lh) {
  union { uint4 q[2]; v16bf v; } u;
  u.q[0] = *(const uint4*)(colp + 8 * lh);
  u.q[1] = *(const uint4*)(colp + 16 + 8 * lh);
  return u.v;
}

// ---------------------------------------------------------------- fp32 -> bf16
__global__ void cast_f32_bf16_kernel(const float* __restrict__ in,
                                     bf16_t* __restrict__ out, int n) {
  int i = blockIdx.x * blockDim.x + threadIdx.x;
  const int stride = gridDim.x * blockDim.x;
  for (; i < n; i += stride) out[i] = f32_to_bf16(in[i]);
}

// ---------------------------------------------------------------- bf16 GEMM
// C(MxN) = A(MxK,row-major) * B(KxN,row-major); 128x128 block, BK=32, 8 waves,
// each wave 32x64 = 2x4 wmma tiles. Double-buffered LDS; A staged via
// async global->LDS (when available), B staged transposed via registers.
#define LDT 48   // LDS pitch in halves (96B, 16B-aligned)

template <bool OUT_BF16>
__global__ void __launch_bounds__(256)
gemm_bf16_kernel(const bf16_t* __restrict__ A, int lda,
                 const bf16_t* __restrict__ B, int ldb,
                 void* __restrict__ Cp, int ldc, int M, int N, int K) {
  __shared__ bf16_t As[2][128 * LDT];
  __shared__ bf16_t Bs[2][128 * LDT];   // transposed: Bs[n][k]

  const int tid = threadIdx.x;
  const int lane = tid & 31;
  const int wave = tid >> 5;
  const int lr = lane & 15, lh = lane >> 4;
  const int wm = (wave & 3) * 32;   // wave M offset
  const int wn = (wave >> 2) * 64;  // wave N offset
  const int blockM = blockIdx.y * 128;
  const int blockN = blockIdx.x * 128;

  const v8f vzero = {0.f, 0.f, 0.f, 0.f, 0.f, 0.f, 0.f, 0.f};
  v8f acc[2][4];
#pragma unroll
  for (int i = 0; i < 2; ++i)
#pragma unroll
    for (int j = 0; j < 4; ++j) acc[i][j] = vzero;

  const int arow = tid >> 1;           // 0..127
  const int acol = (tid & 1) * 16;     // 0 / 16
  const int bn   = tid & 127;          // 0..127
  const int bkh  = (tid >> 7) * 16;    // 0 / 16

  // Out-of-range rows/cols feed accumulators whose stores are guarded, so
  // clamping (instead of zero-fill) is safe for the A path.
  const int grA = (blockM + arow < M) ? (blockM + arow) : (M - 1);

  auto stageA = [&](int buf, int k0) {
    const bf16_t* src = A + (size_t)grA * lda + k0 + acol;
    bf16_t* dst = &As[buf][arow * LDT + acol];
#if HAVE_ASYNC_LDS
    async_g2l_b128(src, dst);          // 16B, ASYNCcnt-tracked, no VGPR trip
    async_g2l_b128(src + 8, dst + 8);
#else
    uint4 d0 = *(const uint4*)(src);
    uint4 d1 = *(const uint4*)(src + 8);
    *(uint4*)dst = d0;
    *(uint4*)(dst + 8) = d1;
#endif
  };
  auto stageB = [&](int buf, int k0) {
    const int gn = blockN + bn;
    bf16_t tmp[16];
    if (gn < N) {
      const bf16_t* src = B + (size_t)(k0 + bkh) * ldb + gn;
#pragma unroll
      for (int i = 0; i < 16; ++i) tmp[i] = src[(size_t)i * ldb];
    } else {
#pragma unroll
      for (int i = 0; i < 16; ++i) tmp[i] = 0;
    }
    *(uint4*)&Bs[buf][bn * LDT + bkh]     = *(uint4*)&tmp[0];
    *(uint4*)&Bs[buf][bn * LDT + bkh + 8] = *(uint4*)&tmp[8];
  };

  const int nk = K >> 5;   // K is a multiple of 32 for every call site
  stageA(0, 0);
  stageB(0, 0);
  wait_async0();
  __syncthreads();

  for (int kt = 0; kt < nk; ++kt) {
    const int cur = kt & 1;
    if (kt + 1 < nk) {         // prefetch next tile into the other buffer
      stageA(cur ^ 1, (kt + 1) * 32);
      stageB(cur ^ 1, (kt + 1) * 32);
    }

    v16bf af[2], bfr[4];
#pragma unroll
    for (int i = 0; i < 2; ++i)
      af[i] = load_afrag(&As[cur][(wm + i * 16 + lr) * LDT], lh);
#pragma unroll
    for (int j = 0; j < 4; ++j)
      bfr[j] = load_bfrag(&Bs[cur][(wn + j * 16 + lr) * LDT], lh);
#pragma unroll
    for (int i = 0; i < 2; ++i)
#pragma unroll
      for (int j = 0; j < 4; ++j) acc[i][j] = wmma_bf16(af[i], bfr[j], acc[i][j]);

    wait_async0();             // own async writes to buf^1 complete
    __syncthreads();           // single barrier per K-step
  }

  // epilogue: C layout (M = r + 8*lh, N = lr)
#pragma unroll
  for (int i = 0; i < 2; ++i) {
#pragma unroll
    for (int j = 0; j < 4; ++j) {
      const int col = blockN + wn + j * 16 + lr;
#pragma unroll
      for (int r = 0; r < 8; ++r) {
        const int row = blockM + wm + i * 16 + r + 8 * lh;
        if (row < M && col < N) {
          const float v = acc[i][j][r];
          if constexpr (OUT_BF16)
            ((bf16_t*)Cp)[(size_t)row * ldc + col] = f32_to_bf16(v);
          else
            ((float*)Cp)[(size_t)row * ldc + col] = v;
        }
      }
    }
  }
}

// ---------------------------------------------------------------- gathers
__global__ void gather_q_kernel(const bf16_t* __restrict__ qfull,
                                bf16_t* __restrict__ qs, int total) {
  int idx = blockIdx.x * 256 + threadIdx.x;
  if (idx >= total) return;
  const int d = idx % QHD;
  const int s = (idx / QHD) % Sc;
  const int h = (idx / (QHD * Sc)) % Hc;
  const int b = idx / (QHD * Sc * Hc);
  qs[idx] = qfull[(size_t)(b * Sc + s) * (Hc * QHD) + h * QHD + d];
}

__global__ void gather_k_kernel(const bf16_t* __restrict__ kv,
                                const bf16_t* __restrict__ cb,
                                bf16_t* __restrict__ ksd, int total) {
  int idx = blockIdx.x * 256 + threadIdx.x;
  if (idx >= total) return;
  const int d = idx % QHD;
  const int s = (idx / QHD) % Sc;
  const int h = (idx / (QHD * Sc)) % Hc;
  const int b = idx / (QHD * Sc * Hc);
  if (d < NOPE)
    ksd[idx] = kv[(size_t)(b * Sc + s) * (Hc * KVUP) + h * KVUP + d];
  else  // shared rope key, broadcast over heads
    ksd[idx] = cb[(size_t)(b * Sc + s) * (KVLORA + ROPE) + KVLORA + (d - NOPE)];
}

__global__ void gather_v_kernel(const bf16_t* __restrict__ kv,
                                bf16_t* __restrict__ vsd, int total) {
  int idx = blockIdx.x * 256 + threadIdx.x;
  if (idx >= total) return;
  const int d = idx % VD;
  const int s = (idx / VD) % Sc;
  const int h = (idx / (VD * Sc)) % Hc;
  const int b = idx / (VD * Sc * Hc);
  vsd[idx] = kv[(size_t)(b * Sc + s) * (Hc * KVUP) + h * KVUP + NOPE + d];
}

// ---------------------------------------------------------------- flash attn
// grid(S/128, B*H), 256 thr (8 waves); wave = 16 q rows; key blocks of 32.
#define KPITCH 208   // 32 x 192 K tile pitch (halves)
#define VPITCH 48    // transposed V / P tile pitch (halves)

__global__ void __launch_bounds__(256)
mla_attn_kernel(const bf16_t* __restrict__ qs, const bf16_t* __restrict__ ks,
                const bf16_t* __restrict__ vs, bf16_t* __restrict__ attn) {
  __shared__ bf16_t Ks[32 * KPITCH];
  __shared__ bf16_t Vt[128 * VPITCH];      // Vt[vd][k]
  __shared__ bf16_t Ps[8 * 16 * VPITCH];   // per-wave 16x32 P tile

  const int tid = threadIdx.x;
  const int lane = tid & 31, wave = tid >> 5;
  const int lr = lane & 15, lh = lane >> 4;
  const int qb = blockIdx.x;
  const int bh = blockIdx.y;
  const int b = bh >> 4, h = bh & 15;
  const int m0 = wave * 16;
  const int qrow0 = qb * 128 + m0;

  // preload Q fragments (16 rows x 192) straight from global (A-layout)
  v16bf qf[6];
  {
    const bf16_t* qrow = qs + ((size_t)bh * Sc + qrow0 + lr) * QHD;
#pragma unroll
    for (int kd = 0; kd < 6; ++kd) qf[kd] = load_afrag(qrow + kd * 32, lh);
  }

  const v8f vzero = {0.f, 0.f, 0.f, 0.f, 0.f, 0.f, 0.f, 0.f};
  v8f o[8];
#pragma unroll
  for (int j = 0; j < 8; ++j) o[j] = vzero;
  float mrow[8], srow[8];
#pragma unroll
  for (int r = 0; r < 8; ++r) { mrow[r] = -3.0e38f; srow[r] = 0.f; }

  const float scale = 0.08838834764831845f;  // 1/sqrt(128)
  const int nkb = (qb + 1) * 4;              // causal: key blocks to process

  for (int kb = 0; kb < nkb; ++kb) {
    __syncthreads();
    {  // stage K block: 32 x 192 row-major; async B128 x3 per thread
      const int krow = tid >> 3;
      const int kcol = (tid & 7) * 24;
      const bf16_t* src = ks + ((size_t)bh * Sc + kb * 32 + krow) * QHD + kcol;
      bf16_t* dst = &Ks[krow * KPITCH + kcol];
#if HAVE_ASYNC_LDS
      async_g2l_b128(src, dst);
      async_g2l_b128(src + 8, dst + 8);
      async_g2l_b128(src + 16, dst + 16);
#else
      uint4 d0 = *(const uint4*)(src);
      uint4 d1 = *(const uint4*)(src + 8);
      uint4 d2 = *(const uint4*)(src + 16);
      *(uint4*)(dst) = d0;
      *(uint4*)(dst + 8) = d1;
      *(uint4*)(dst + 16) = d2;
#endif
    }
    {  // stage V block transposed: Vt[vd][k] (scalar gather, coalesced loads)
      const int vd = tid & 127;
      const int kh2 = (tid >> 7) * 16;
      const bf16_t* src = vs + ((size_t)bh * Sc + kb * 32 + kh2) * VD + vd;
      bf16_t tmp[16];
#pragma unroll
      for (int i = 0; i < 16; ++i) tmp[i] = src[(size_t)i * VD];
      *(uint4*)&Vt[vd * VPITCH + kh2]     = *(uint4*)&tmp[0];
      *(uint4*)&Vt[vd * VPITCH + kh2 + 8] = *(uint4*)&tmp[8];
    }
    wait_async0();
    __syncthreads();

    const bool active = (kb * 32) <= (qrow0 + 15);  // wave-uniform causal skip
    if (active) {
      // S = Q * K^T (16x32), K columns are K rows => K-contiguous B frags
      v8f sc0 = vzero, sc1 = vzero;
#pragma unroll
      for (int kd = 0; kd < 6; ++kd) {
        v16bf k0f = load_bfrag(&Ks[lr * KPITCH + kd * 32], lh);
        v16bf k1f = load_bfrag(&Ks[(16 + lr) * KPITCH + kd * 32], lh);
        sc0 = wmma_bf16(qf[kd], k0f, sc0);
        sc1 = wmma_bf16(qf[kd], k1f, sc1);
      }
      // scale + causal mask (C layout: row = qrow0 + r + 8*lh, col = lr)
      const int rbase = qrow0 + 8 * lh;
#pragma unroll
      for (int r = 0; r < 8; ++r) {
        const int qi = rbase + r;
        const float s0 = sc0[r] * scale;
        const float s1 = sc1[r] * scale;
        sc0[r] = ((kb * 32 + lr) <= qi) ? s0 : -3.0e38f;
        sc1[r] = ((kb * 32 + 16 + lr) <= qi) ? s1 : -3.0e38f;
      }
      // online softmax: reductions across the 16-lane half-wave
#pragma unroll
      for (int r = 0; r < 8; ++r) {
        float mx = fmaxf(sc0[r], sc1[r]);
        mx = fmaxf(mx, __shfl_xor(mx, 1, 32));
        mx = fmaxf(mx, __shfl_xor(mx, 2, 32));
        mx = fmaxf(mx, __shfl_xor(mx, 4, 32));
        mx = fmaxf(mx, __shfl_xor(mx, 8, 32));
        const float mnew = fmaxf(mrow[r], mx);
        const float alpha = __expf(mrow[r] - mnew);
        const float p0 = __expf(sc0[r] - mnew);
        const float p1 = __expf(sc1[r] - mnew);
        sc0[r] = p0;
        sc1[r] = p1;
        float ps = p0 + p1;
        ps += __shfl_xor(ps, 1, 32);
        ps += __shfl_xor(ps, 2, 32);
        ps += __shfl_xor(ps, 4, 32);
        ps += __shfl_xor(ps, 8, 32);
        srow[r] = srow[r] * alpha + ps;
        mrow[r] = mnew;
#pragma unroll
        for (int j = 0; j < 8; ++j) o[j][r] *= alpha;
      }
      // P (C-layout) -> wave-private LDS -> A-fragment, then O += P*V
      bf16_t* pw = &Ps[wave * 16 * VPITCH];
#pragma unroll
      for (int r = 0; r < 8; ++r) {
        pw[(r + 8 * lh) * VPITCH + lr]      = f32_to_bf16(sc0[r]);
        pw[(r + 8 * lh) * VPITCH + 16 + lr] = f32_to_bf16(sc1[r]);
      }
      asm volatile("s_wait_dscnt 0x0" ::: "memory");  // wave-local store->load
      const v16bf pf = load_afrag(&pw[lr * VPITCH], lh);
#pragma unroll
      for (int j = 0; j < 8; ++j) {
        const v16bf vf = load_bfrag(&Vt[(j * 16 + lr) * VPITCH], lh);
        o[j] = wmma_bf16(pf, vf, o[j]);
      }
    }
  }

  // normalize and store to (B, S, H*VD) bf16 for the final GEMM
#pragma unroll
  for (int j = 0; j < 8; ++j) {
#pragma unroll
    for (int r = 0; r < 8; ++r) {
      const int row = qrow0 + r + 8 * lh;
      const float v = o[j][r] / srow[r];
      attn[((size_t)b * Sc + row) * (Hc * VD) + h * VD + j * 16 + lr] =
          f32_to_bf16(v);
    }
  }
}

// ---------------------------------------------------------------- launcher
extern "C" void kernel_launch(void* const* d_in, const int* in_sizes, int n_in,
                              void* d_out, int out_size, void* d_ws,
                              size_t ws_size, hipStream_t stream) {
  (void)in_sizes; (void)n_in; (void)out_size; (void)ws_size;
  const float* x         = (const float*)d_in[0];
  // d_in[1] = position (unused by reference math), d_in[2] = mask (causal tril)
  const float* w_q_down  = (const float*)d_in[3];
  const float* w_q_up    = (const float*)d_in[4];
  const float* w_kv_down = (const float*)d_in[5];
  const float* w_kv_up   = (const float*)d_in[6];
  const float* w_out     = (const float*)d_in[7];
  float* out = (float*)d_out;

  char* ws = (char*)d_ws;
  size_t off = 0;
  auto alloc = [&](size_t elems) -> bf16_t* {
    bf16_t* p = (bf16_t*)(ws + off);
    off += ((elems * sizeof(bf16_t) + 255) & ~(size_t)255);
    return p;
  };
  bf16_t* xb    = alloc((size_t)MROWS * Dc);
  bf16_t* wqd   = alloc((size_t)Dc * QLORA);
  bf16_t* wqu   = alloc((size_t)QLORA * (Hc * QHD));
  bf16_t* wkvd  = alloc((size_t)Dc * (KVLORA + ROPE));
  bf16_t* wkvu  = alloc((size_t)KVLORA * (Hc * KVUP));
  bf16_t* woutb = alloc((size_t)(Hc * VD) * Dc);
  bf16_t* qd    = alloc((size_t)MROWS * QLORA);
  bf16_t* qfull = alloc((size_t)MROWS * (Hc * QHD));
  bf16_t* cb    = alloc((size_t)MROWS * (KVLORA + ROPE));
  bf16_t* kvb   = alloc((size_t)MROWS * (Hc * KVUP));
  bf16_t* qsb   = alloc((size_t)Bc * Hc * Sc * QHD);
  bf16_t* ksb   = alloc((size_t)Bc * Hc * Sc * QHD);
  bf16_t* vsb   = alloc((size_t)Bc * Hc * Sc * VD);
  bf16_t* attn  = alloc((size_t)MROWS * (Hc * VD));

  auto cast = [&](const float* src, bf16_t* dst, size_t n) {
    int blocks = (int)((n + 255) / 256);
    if (blocks > 4096) blocks = 4096;
    cast_f32_bf16_kernel<<<blocks, 256, 0, stream>>>(src, dst, (int)n);
  };
  cast(x, xb, (size_t)MROWS * Dc);
  cast(w_q_down, wqd, (size_t)Dc * QLORA);
  cast(w_q_up, wqu, (size_t)QLORA * (Hc * QHD));
  cast(w_kv_down, wkvd, (size_t)Dc * (KVLORA + ROPE));
  cast(w_kv_up, wkvu, (size_t)KVLORA * (Hc * KVUP));
  cast(w_out, woutb, (size_t)(Hc * VD) * Dc);

  auto gemm_b = [&](const bf16_t* A, int lda, const bf16_t* B2, int ldb,
                    bf16_t* C, int ldc, int M, int N, int K) {
    dim3 g((N + 127) / 128, (M + 127) / 128);
    gemm_bf16_kernel<true><<<g, 256, 0, stream>>>(A, lda, B2, ldb, (void*)C,
                                                  ldc, M, N, K);
  };
  // q_down = x @ w_q_down                       (4096x2048)(2048x768)
  gemm_b(xb, Dc, wqd, QLORA, qd, QLORA, MROWS, QLORA, Dc);
  // q = q_down @ w_q_up                         (4096x768)(768x3072)
  gemm_b(qd, QLORA, wqu, Hc * QHD, qfull, Hc * QHD, MROWS, Hc * QHD, QLORA);
  // c = x @ w_kv_down                           (4096x2048)(2048x576)
  gemm_b(xb, Dc, wkvd, KVLORA + ROPE, cb, KVLORA + ROPE, MROWS, KVLORA + ROPE, Dc);
  // kv = c[:, :512] @ w_kv_up                   (4096x512)(512x4096)
  gemm_b(cb, KVLORA + ROPE, wkvu, Hc * KVUP, kvb, Hc * KVUP, MROWS, Hc * KVUP,
         KVLORA);

  const int tq = Bc * Hc * Sc * QHD;
  gather_q_kernel<<<(tq + 255) / 256, 256, 0, stream>>>(qfull, qsb, tq);
  gather_k_kernel<<<(tq + 255) / 256, 256, 0, stream>>>(kvb, cb, ksb, tq);
  const int tv = Bc * Hc * Sc * VD;
  gather_v_kernel<<<(tv + 255) / 256, 256, 0, stream>>>(kvb, vsb, tv);

  dim3 ag(Sc / 128, Bc * Hc);
  mla_attn_kernel<<<ag, 256, 0, stream>>>(qsb, ksb, vsb, attn);

  // out = attn @ w_out (fp32 output)            (4096x2048)(2048x2048)
  dim3 g2((Dc + 127) / 128, (MROWS + 127) / 128);
  gemm_bf16_kernel<false><<<g2, 256, 0, stream>>>(attn, Hc * VD, woutb, Dc,
                                                  (void*)out, Dc, MROWS, Dc,
                                                  Hc * VD);
}
